// EquiformerEncoder_23356032155889
// MI455X (gfx1250) — matrix operand order repaired and checked by reading, not compiled, not run on hardware
//
#include <hip/hip_runtime.h>
#include <hip/hip_bf16.h>

// ---------------------------------------------------------------------------
// EquiformerEncoder forward for gfx1250 (MI455X). All dense math runs on
// v_wmma_f32_16x16x32_f16 (f16 inputs, f32 accumulate). The big [E,49,128]
// "m"/"v" intermediates are never written to HBM: a fused per-edge kernel
// builds m in LDS, does the 49x128 @ 128x128 value GEMM with WMMA (one head
// per wave), scales by alpha and scatter-adds with f32 global atomics.
// B operands are staged TRANSPOSED [n][k] in LDS so both A and B fragments
// read as contiguous 32-byte chunks (ds_load_b128), no scalar repacking.
// ---------------------------------------------------------------------------

#define KSH   49
#define CCH   128
#define NRBF_ 512
#define FEATK 800          // 774 padded to multiple of 32
#define RADN  896          // 7*128
#define NLAY  12
#define FFNF  512
#define NODE_ELT (KSH*CCH) // 6272

typedef __attribute__((ext_vector_type(16))) _Float16 v16h;
typedef __attribute__((ext_vector_type(8)))  float    v8f;

__device__ const int g_LIDX[49] = {0,1,1,1,2,2,2,2,2,3,3,3,3,3,3,3,
                                   4,4,4,4,4,4,4,4,4,
                                   5,5,5,5,5,5,5,5,5,5,5,
                                   6,6,6,6,6,6,6,6,6,6,6,6,6};
__device__ const int g_M0[7] = {0,2,6,12,20,30,42};

#define CDIV(a,b) (((a)+(b)-1)/(b))

// ---------------- WMMA fragment loaders (16x16x32 f16, contiguous k order in A and B) --------
// A (16x32, MxK) in LDS as [m][k]: lane l -> m=l&15, g=l>>4;
//   half i -> k = g*16 + i  => one contiguous 16-half run.
__device__ __forceinline__ v16h load_frag_a(const _Float16* sm, int ld) {
  const int lane = threadIdx.x & 31;
  const int m = lane & 15, g = lane >> 4;
  v16h f;
#pragma unroll
  for (int i = 0; i < 16; ++i) {
    int k = (g << 4) + i;
    f[i] = sm[m * ld + k];
  }
  return f;
}
// B (32x16, KxN) staged TRANSPOSED in LDS as [n][k]: lane l -> n=l&15, g=l>>4;
//   half i -> k = g*16 + i  => one contiguous 16-half (32B) run per lane.
__device__ __forceinline__ v16h load_frag_bt(const _Float16* sm, int ld) {
  const int lane = threadIdx.x & 31;
  const int n = lane & 15, g = lane >> 4;
  v16h f;
#pragma unroll
  for (int i = 0; i < 16; ++i) {
    f[i] = sm[n * ld + (g << 4) + i];
  }
  return f;
}

// ---------------- Generic WMMA GEMM: C[M,N] = act(A[M,K]@B[K,N]) -----------
// Block: 256 threads = 8 waves -> 64(M) x 64(N) tile. Wave w owns M-tile
// (w>>1) and N-tiles (w&1)*2 + {0,1}, sharing one A fragment per K-step.
template<int BIAS, int SILU, int GATE, int STF16, int STF32, int ACC>
__global__ __launch_bounds__(256)
void gemm16(const _Float16* __restrict__ A, int lda,
            const _Float16* __restrict__ B, int ldb,
            const float* __restrict__ bias,
            const float* __restrict__ gate, int gate_div, int gate_ld,
            float* __restrict__ Cf, _Float16* __restrict__ Ch, int ldc,
            int M, int Kd, int Nd)
{
  __shared__ _Float16 As[64][40];    // [m][k], row stride 80B (16B aligned)
  __shared__ _Float16 Bt[64][40];    // [n][k] transposed
  const int tid  = threadIdx.x;
  const int wave = tid >> 5;
  const int mt = wave >> 1;          // 0..3
  const int nh = wave & 1;           // 0..1 -> N tiles nh*2, nh*2+1
  const int m0 = blockIdx.x * 64;
  const int n0 = blockIdx.y * 64;
  v8f acc0 = {}, acc1 = {};
  for (int k0 = 0; k0 < Kd; k0 += 32) {
    for (int i = tid; i < 64 * 32; i += 256) {
      int r = i >> 5, c = i & 31;
      int gr = m0 + r, gk = k0 + c;
      As[r][c] = (gr < M && gk < Kd) ? A[(size_t)gr * lda + gk] : (_Float16)0.f;
    }
    for (int i = tid; i < 32 * 64; i += 256) {
      int r = i >> 6, c = i & 63;              // r = k, c = n (coalesced read)
      int gk = k0 + r, gn = n0 + c;
      Bt[c][r] = (gk < Kd && gn < Nd) ? B[(size_t)gk * ldb + gn] : (_Float16)0.f;
    }
    __syncthreads();
    v16h a  = load_frag_a (&As[mt * 16][0], 40);
    v16h b0 = load_frag_bt(&Bt[(nh * 2 + 0) * 16][0], 40);
    v16h b1 = load_frag_bt(&Bt[(nh * 2 + 1) * 16][0], 40);
    acc0 = __builtin_amdgcn_wmma_f32_16x16x32_f16(false, a, false, b0,
                                                  (short)0, acc0, false, false);
    acc1 = __builtin_amdgcn_wmma_f32_16x16x32_f16(false, a, false, b1,
                                                  (short)0, acc1, false, false);
    __syncthreads();
  }
  const int lane = tid & 31, nn = lane & 15, gg = lane >> 4;
#pragma unroll
  for (int t = 0; t < 2; ++t) {
    const int col = n0 + (nh * 2 + t) * 16 + nn;
    if (col >= Nd) continue;
#pragma unroll
    for (int r = 0; r < 8; ++r) {
      int row = m0 + mt * 16 + r + gg * 8;
      if (row >= M) continue;
      float v = t ? acc1[r] : acc0[r];
      if (BIAS) v += bias[col];
      if (SILU) v = v / (1.f + __expf(-v));
      if (GATE) v *= gate[(size_t)(row / gate_div) * gate_ld + col];
      if (ACC)        Cf[(size_t)row * ldc + col] += v;
      else if (STF32) Cf[(size_t)row * ldc + col]  = v;
      if (STF16)      Ch[(size_t)row * ldc + col]  = (_Float16)v;
    }
  }
}

// ---------------- Fused per-edge attention value GEMM ----------------------
// One block per edge: m = (xn[src]+xn[dst])*rad[L(k)] built in LDS (f16),
// Wv staged transposed in LDS, wave w == head w keeps its 4 K-step B
// fragments in registers and streams 4 M-tiles of A, scales by alpha,
// atomic-adds into agg[dst].
__global__ __launch_bounds__(256)
void attn_edge_k(const float* __restrict__ xn, const float* __restrict__ rad,
                 const _Float16* __restrict__ Wv, const float* __restrict__ alpha,
                 const int* __restrict__ ei, float* __restrict__ agg, int nE)
{
  const int e = blockIdx.x;
  if (e >= nE) return;
  __shared__ _Float16 Ms[64][136];    // [k_sh row m][c] (A operand, [m][k])
  __shared__ _Float16 Wt[128][136];   // Wv transposed: [n][k]
  const int tid = threadIdx.x;
  const int src = ei[e], dst = ei[nE + e];
  for (int i = tid; i < 128 * 128; i += 256) {
    int r = i >> 7, c = i & 127;      // r = k (row of Wv), c = n
    Wt[c][r] = Wv[i];
  }
  for (int i = tid; i < 64 * 128; i += 256) {
    int k = i >> 7, c = i & 127;
    float v = 0.f;
    if (k < KSH) {
      int l = g_LIDX[k];
      v = (xn[(size_t)src * NODE_ELT + k * 128 + c] +
           xn[(size_t)dst * NODE_ELT + k * 128 + c]) *
          rad[(size_t)e * RADN + l * 128 + c];
    }
    Ms[k][c] = (_Float16)v;
  }
  __syncthreads();
  const int head = tid >> 5;                 // wave id == head
  const float al = alpha[(size_t)e * 8 + head];
  const int lane = tid & 31, nn = lane & 15, gg = lane >> 4;
  v16h bf[4];
#pragma unroll
  for (int ks = 0; ks < 4; ++ks)
    bf[ks] = load_frag_bt(&Wt[head * 16][ks * 32], 136);
  for (int mt = 0; mt < 4; ++mt) {
    v8f acc = {};
#pragma unroll
    for (int ks = 0; ks < 4; ++ks) {
      v16h a = load_frag_a(&Ms[mt * 16][ks * 32], 136);
      acc = __builtin_amdgcn_wmma_f32_16x16x32_f16(false, a, false, bf[ks],
                                                   (short)0, acc, false, false);
    }
#pragma unroll
    for (int r = 0; r < 8; ++r) {
      int row = mt * 16 + r + gg * 8;
      if (row < KSH)
        atomicAdd(&agg[(size_t)dst * NODE_ELT + row * 128 + head * 16 + nn],
                  acc[r] * al);
    }
  }
}

// ---------------- RMS norm over (m, channel) per degree --------------------
__global__ __launch_bounds__(128)
void rmsnorm_k(const float* __restrict__ x, const float* __restrict__ w,
               float* __restrict__ yf, _Float16* __restrict__ yh, int nN)
{
  constexpr int LID[49] = {0,1,1,1,2,2,2,2,2,3,3,3,3,3,3,3,
                           4,4,4,4,4,4,4,4,4,
                           5,5,5,5,5,5,5,5,5,5,5,
                           6,6,6,6,6,6,6,6,6,6,6,6,6};
  const int n = blockIdx.x;
  if (n >= nN) return;
  __shared__ float ssum[8];
  const int t = threadIdx.x;
  if (t < 8) ssum[t] = 0.f;
  __syncthreads();
  const float* xr = x + (size_t)n * NODE_ELT;
  float p[7] = {0, 0, 0, 0, 0, 0, 0};
#pragma unroll
  for (int k = 0; k < 49; ++k) { float v = xr[k * 128 + t]; p[LID[k]] += v * v; }
#pragma unroll
  for (int l = 0; l < 7; ++l) atomicAdd(&ssum[l], p[l]);
  __syncthreads();
  float inv[7];
#pragma unroll
  for (int l = 0; l < 7; ++l)
    inv[l] = rsqrtf(ssum[l] / ((2 * l + 1) * 128.0f) + 1e-6f);
  float* yr = yf + (size_t)n * NODE_ELT;
#pragma unroll
  for (int k = 0; k < 49; ++k) {
    int l = LID[k];
    float v = xr[k * 128 + t] * inv[l] * w[l * 128 + t];
    yr[k * 128 + t] = v;
    if (yh) yh[(size_t)n * NODE_ELT + k * 128 + t] = (_Float16)v;
  }
}

// ---------------- Small helper kernels -------------------------------------
__global__ void init_x_k(const int* __restrict__ z, const float* __restrict__ frac,
                         const float* __restrict__ aemb, const float* __restrict__ fw,
                         float* __restrict__ x, int nN)
{
  int idx = blockIdx.x * blockDim.x + threadIdx.x;
  if (idx >= nN * NODE_ELT) return;
  int n = idx / NODE_ELT, r = idx % NODE_ELT, k = r >> 7, c = r & 127;
  float v = 0.f;
  if (k == 0)
    v = aemb[z[n] * 128 + c] + frac[n * 3 + 0] * fw[c] +
        frac[n * 3 + 1] * fw[128 + c] + frac[n * 3 + 2] * fw[256 + c];
  x[idx] = v;
}

__global__ void edge_scalar_k(const float* __restrict__ d, const float* __restrict__ ev,
                              const float* __restrict__ efd, _Float16* __restrict__ feat,
                              int nE)
{
  long long idx = (long long)blockIdx.x * blockDim.x + threadIdx.x;
  if (idx >= (long long)nE * FEATK) return;
  int e = (int)(idx / FEATK), j = (int)(idx % FEATK);
  const float step = 5.0f / 511.0f;
  const float coeff = -0.5f / ((2.0f * step) * (2.0f * step));
  float v;
  if (j < 512)      { float df = d[e] - j * step; v = __expf(coeff * df * df); }
  else if (j < 515)   v = ev[e * 3 + (j - 512)];
  else if (j < 518)   v = efd[e * 3 + (j - 515)];
  else if (j < 774)   return;          // layer embeddings fill this region
  else                v = 0.f;         // K padding
  feat[(size_t)e * FEATK + j] = (_Float16)v;
}

__global__ void fill_emb_k(const int* __restrict__ ei, const int* __restrict__ z,
                           const float* __restrict__ semb, const float* __restrict__ temb,
                           _Float16* __restrict__ feat, int nE)
{
  long long idx = (long long)blockIdx.x * blockDim.x + threadIdx.x;
  if (idx >= (long long)nE * 256) return;
  int e = (int)(idx / 256), j = (int)(idx % 256);
  int node = (j < 128) ? ei[e] : ei[nE + e];
  int zz = z[node];
  float v = (j < 128) ? semb[zz * 128 + j] : temb[zz * 128 + (j - 128)];
  feat[(size_t)e * FEATK + 518 + j] = (_Float16)v;
}

__global__ void cvt_pad_k(const float* __restrict__ src, _Float16* __restrict__ dst,
                          int rows, int cols, int drows)
{
  long long idx = (long long)blockIdx.x * blockDim.x + threadIdx.x;
  if (idx >= (long long)drows * cols) return;
  int r = (int)(idx / cols);
  dst[idx] = (r < rows) ? (_Float16)src[idx] : (_Float16)0.f;
}

__global__ void deg_scatter_k(const float* __restrict__ rad, const int* __restrict__ ei,
                              float* __restrict__ x, int nE)
{
  long long idx = (long long)blockIdx.x * blockDim.x + threadIdx.x;
  if (idx >= (long long)nE * RADN) return;
  int e = (int)(idx / RADN), r = (int)(idx % RADN), l = r >> 7, c = r & 127;
  atomicAdd(&x[(size_t)ei[nE + e] * NODE_ELT + g_M0[l] * 128 + c], rad[idx]);
}

__global__ void m0_k(const float* __restrict__ xn, const float* __restrict__ rad,
                     const int* __restrict__ ei, _Float16* __restrict__ m0, int nE)
{
  long long idx = (long long)blockIdx.x * blockDim.x + threadIdx.x;
  if (idx >= (long long)nE * 128) return;
  int e = (int)(idx >> 7), c = (int)(idx & 127);
  int src = ei[e], dst = ei[nE + e];
  float v = (xn[(size_t)src * NODE_ELT + c] + xn[(size_t)dst * NODE_ELT + c]) *
            rad[(size_t)e * RADN + c];
  m0[idx] = (_Float16)v;
}

__device__ __forceinline__ unsigned f_enc(float f) {
  unsigned u = __float_as_uint(f);
  return (u >> 31) ? ~u : (u | 0x80000000u);
}
__device__ __forceinline__ float f_dec(unsigned u) {
  return __uint_as_float((u >> 31) ? (u & 0x7FFFFFFFu) : ~u);
}

__global__ void logits_k(const float* __restrict__ a, const float* __restrict__ va,
                         const int* __restrict__ ei, float* __restrict__ lg,
                         unsigned* __restrict__ mxe, int nE)
{
  int idx = blockIdx.x * blockDim.x + threadIdx.x;
  if (idx >= nE * 8) return;
  int e = idx >> 3, h = idx & 7;
  float s = 0.f;
#pragma unroll
  for (int j = 0; j < 32; ++j) s += a[(size_t)e * 256 + h * 32 + j] * va[h * 32 + j];
  lg[idx] = s;
  atomicMax(&mxe[ei[nE + e] * 8 + h], f_enc(s));
}

__global__ void expsum_k(float* __restrict__ lg, const unsigned* __restrict__ mxe,
                         const int* __restrict__ ei, float* __restrict__ den, int nE)
{
  int idx = blockIdx.x * blockDim.x + threadIdx.x;
  if (idx >= nE * 8) return;
  int e = idx >> 3, h = idx & 7;
  float ex = __expf(lg[idx] - f_dec(mxe[ei[nE + e] * 8 + h]));
  lg[idx] = ex;
  atomicAdd(&den[ei[nE + e] * 8 + h], ex);
}

__global__ void alpha_k(float* __restrict__ lg, const float* __restrict__ den,
                        const int* __restrict__ ei, int nE)
{
  int idx = blockIdx.x * blockDim.x + threadIdx.x;
  if (idx >= nE * 8) return;
  int e = idx >> 3, h = idx & 7;
  lg[idx] = lg[idx] / (den[ei[nE + e] * 8 + h] + 1e-9f);
}

// ---------------------------------------------------------------------------
extern "C" void kernel_launch(void* const* d_in, const int* in_sizes, int n_in,
                              void* d_out, int out_size, void* d_ws, size_t ws_size,
                              hipStream_t stream)
{
  // Positional inputs (setup_inputs order), then params flattened in jax
  // pytree order (dict keys sorted alphabetically; 'layers' nested, keys
  // W1,W2,Wa,Wg,Wo,Wv,n1,n2,rb1,rb2,rw1,rw2,rw3,src_emb,tgt_emb,va).
  const int*   z     = (const int*)  d_in[0];
  const float* frac  = (const float*)d_in[1];
  const int*   ei    = (const int*)  d_in[2];
  const float* dist  = (const float*)d_in[3];
  const float* evec  = (const float*)d_in[4];
  const float* efd   = (const float*)d_in[5];
  const float* atom_emb = (const float*)d_in[6];
  const float* db1   = (const float*)d_in[7];
  const float* db2   = (const float*)d_in[8];
  const float* deg_s = (const float*)d_in[9];
  const float* deg_t = (const float*)d_in[10];
  const float* dw1   = (const float*)d_in[11];
  const float* dw2   = (const float*)d_in[12];
  const float* dw3   = (const float*)d_in[13];
  const float* fracW = (const float*)d_in[14];
  const float* L_W1  = (const float*)d_in[15];
  const float* L_W2  = (const float*)d_in[16];
  const float* L_Wa  = (const float*)d_in[17];
  const float* L_Wg  = (const float*)d_in[18];
  const float* L_Wo  = (const float*)d_in[19];
  const float* L_Wv  = (const float*)d_in[20];
  const float* L_n1  = (const float*)d_in[21];
  const float* L_n2  = (const float*)d_in[22];
  const float* L_rb1 = (const float*)d_in[23];
  const float* L_rb2 = (const float*)d_in[24];
  const float* L_rw1 = (const float*)d_in[25];
  const float* L_rw2 = (const float*)d_in[26];
  const float* L_rw3 = (const float*)d_in[27];
  const float* L_se  = (const float*)d_in[28];
  const float* L_te  = (const float*)d_in[29];
  const float* L_va  = (const float*)d_in[30];
  const float* nf    = (const float*)d_in[31];

  const int nN = in_sizes[0];
  const int nE = in_sizes[2] / 2;
  float* x = (float*)d_out;                 // running state lives in d_out

  // ---- workspace arena ----
  char* p = (char*)d_ws;
  auto arena = [&](size_t bytes) -> void* {
    void* r = (void*)p; p += (bytes + 255) & ~(size_t)255; return r;
  };
  _Float16* feat = (_Float16*)arena((size_t)nE * FEATK * 2);
  _Float16* h1   = (_Float16*)arena((size_t)nE * 128 * 2);
  _Float16* h2   = (_Float16*)arena((size_t)nE * 128 * 2);
  float*    rad  = (float*)   arena((size_t)nE * RADN * 4);
  float*    xnf  = (float*)   arena((size_t)nN * NODE_ELT * 4);
  _Float16* xnh  = (_Float16*)arena((size_t)nN * NODE_ELT * 2);
  float*    agg  = (float*)   arena((size_t)nN * NODE_ELT * 4);
  _Float16* aggh = (_Float16*)arena((size_t)nN * NODE_ELT * 2);
  _Float16* m0b  = (_Float16*)arena((size_t)nE * 128 * 2);
  float*    abuf = (float*)   arena((size_t)nE * 256 * 4);
  float*    attw = (float*)   arena((size_t)nE * 8 * 4);
  unsigned* mxe  = (unsigned*)arena((size_t)nN * 8 * 4);
  float*    den  = (float*)   arena((size_t)nN * 8 * 4);
  float*    gate = (float*)   arena((size_t)nN * FFNF * 4);
  _Float16* hid  = (_Float16*)arena((size_t)nN * KSH * FFNF * 2);
  _Float16* w1p  = (_Float16*)arena((size_t)FEATK * 128 * 2);
  _Float16* w2h  = (_Float16*)arena((size_t)128 * 128 * 2);
  _Float16* w3h  = (_Float16*)arena((size_t)128 * RADN * 2);
  _Float16* wah  = (_Float16*)arena((size_t)128 * 256 * 2);
  _Float16* wvh  = (_Float16*)arena((size_t)128 * 128 * 2);
  _Float16* woh  = (_Float16*)arena((size_t)128 * 128 * 2);
  _Float16* wgh  = (_Float16*)arena((size_t)128 * FFNF * 2);
  _Float16* wf1  = (_Float16*)arena((size_t)128 * FFNF * 2);
  _Float16* wf2  = (_Float16*)arena((size_t)FFNF * 128 * 2);
  (void)ws_size; (void)n_in; (void)out_size;

  // ---- once: initial embedding + invariant edge scalars ----
  init_x_k<<<CDIV(nN * NODE_ELT, 256), 256, 0, stream>>>(z, frac, atom_emb, fracW, x, nN);
  edge_scalar_k<<<CDIV((long long)nE * FEATK, 256), 256, 0, stream>>>(dist, evec, efd, feat, nE);

  // ---- once: edge-degree embedding (radial MLP on degree features) ----
  fill_emb_k<<<CDIV(nE * 256, 256), 256, 0, stream>>>(ei, z, deg_s, deg_t, feat, nE);
  cvt_pad_k<<<CDIV(FEATK * 128, 256), 256, 0, stream>>>(dw1, w1p, 774, 128, FEATK);
  cvt_pad_k<<<CDIV(128 * 128, 256), 256, 0, stream>>>(dw2, w2h, 128, 128, 128);
  cvt_pad_k<<<CDIV(128 * RADN, 256), 256, 0, stream>>>(dw3, w3h, 128, RADN, 128);
  gemm16<1,1,0,1,0,0><<<dim3(CDIV(nE,64), 2), 256, 0, stream>>>(
      feat, FEATK, w1p, 128, db1, nullptr, 1, 1, nullptr, h1, 128, nE, FEATK, 128);
  gemm16<1,1,0,1,0,0><<<dim3(CDIV(nE,64), 2), 256, 0, stream>>>(
      h1, 128, w2h, 128, db2, nullptr, 1, 1, nullptr, h2, 128, nE, 128, 128);
  gemm16<0,0,0,0,1,0><<<dim3(CDIV(nE,64), RADN/64), 256, 0, stream>>>(
      h2, 128, w3h, RADN, nullptr, nullptr, 1, 1, rad, nullptr, RADN, nE, 128, RADN);
  deg_scatter_k<<<CDIV((long long)nE * RADN, 256), 256, 0, stream>>>(rad, ei, x, nE);

  // ---- 12 transformer blocks ----
  for (int L = 0; L < NLAY; ++L) {
    const float* n1  = L_n1  + (size_t)L * 7 * 128;
    const float* n2  = L_n2  + (size_t)L * 7 * 128;
    const float* se  = L_se  + (size_t)L * 90 * 128;
    const float* te  = L_te  + (size_t)L * 90 * 128;
    const float* rw1 = L_rw1 + (size_t)L * 774 * 128;
    const float* rw2 = L_rw2 + (size_t)L * 128 * 128;
    const float* rw3 = L_rw3 + (size_t)L * 128 * RADN;
    const float* rb1 = L_rb1 + (size_t)L * 128;
    const float* rb2 = L_rb2 + (size_t)L * 128;
    const float* Wa  = L_Wa  + (size_t)L * 128 * 256;
    const float* va  = L_va  + (size_t)L * 8 * 32;
    const float* Wv  = L_Wv  + (size_t)L * 128 * 128;
    const float* Wo  = L_Wo  + (size_t)L * 128 * 128;
    const float* W1  = L_W1  + (size_t)L * 128 * FFNF;
    const float* Wg  = L_Wg  + (size_t)L * 128 * FFNF;
    const float* W2  = L_W2  + (size_t)L * FFNF * 128;

    // --- attention half ---
    rmsnorm_k<<<nN, 128, 0, stream>>>(x, n1, xnf, xnh, nN);
    fill_emb_k<<<CDIV(nE * 256, 256), 256, 0, stream>>>(ei, z, se, te, feat, nE);
    cvt_pad_k<<<CDIV(FEATK * 128, 256), 256, 0, stream>>>(rw1, w1p, 774, 128, FEATK);
    cvt_pad_k<<<CDIV(128 * 128, 256), 256, 0, stream>>>(rw2, w2h, 128, 128, 128);
    cvt_pad_k<<<CDIV(128 * RADN, 256), 256, 0, stream>>>(rw3, w3h, 128, RADN, 128);
    cvt_pad_k<<<CDIV(128 * 256, 256), 256, 0, stream>>>(Wa, wah, 128, 256, 128);
    cvt_pad_k<<<CDIV(128 * 128, 256), 256, 0, stream>>>(Wv, wvh, 128, 128, 128);
    cvt_pad_k<<<CDIV(128 * 128, 256), 256, 0, stream>>>(Wo, woh, 128, 128, 128);
    cvt_pad_k<<<CDIV(128 * FFNF, 256), 256, 0, stream>>>(Wg, wgh, 128, FFNF, 128);
    cvt_pad_k<<<CDIV(128 * FFNF, 256), 256, 0, stream>>>(W1, wf1, 128, FFNF, 128);
    cvt_pad_k<<<CDIV(FFNF * 128, 256), 256, 0, stream>>>(W2, wf2, FFNF, 128, FFNF);

    gemm16<1,1,0,1,0,0><<<dim3(CDIV(nE,64), 2), 256, 0, stream>>>(
        feat, FEATK, w1p, 128, rb1, nullptr, 1, 1, nullptr, h1, 128, nE, FEATK, 128);
    gemm16<1,1,0,1,0,0><<<dim3(CDIV(nE,64), 2), 256, 0, stream>>>(
        h1, 128, w2h, 128, rb2, nullptr, 1, 1, nullptr, h2, 128, nE, 128, 128);
    gemm16<0,0,0,0,1,0><<<dim3(CDIV(nE,64), RADN/64), 256, 0, stream>>>(
        h2, 128, w3h, RADN, nullptr, nullptr, 1, 1, rad, nullptr, RADN, nE, 128, RADN);

    m0_k<<<CDIV((long long)nE * 128, 256), 256, 0, stream>>>(xnf, rad, ei, m0b, nE);
    gemm16<0,1,0,0,1,0><<<dim3(CDIV(nE,64), 4), 256, 0, stream>>>(
        m0b, 128, wah, 256, nullptr, nullptr, 1, 1, abuf, nullptr, 256, nE, 128, 256);

    hipMemsetAsync(mxe, 0, (size_t)nN * 8 * 4, stream);
    hipMemsetAsync(den, 0, (size_t)nN * 8 * 4, stream);
    logits_k<<<CDIV(nE * 8, 256), 256, 0, stream>>>(abuf, va, ei, attw, mxe, nE);
    expsum_k<<<CDIV(nE * 8, 256), 256, 0, stream>>>(attw, mxe, ei, den, nE);
    alpha_k<<<CDIV(nE * 8, 256), 256, 0, stream>>>(attw, den, ei, nE);

    hipMemsetAsync(agg, 0, (size_t)nN * NODE_ELT * 4, stream);
    attn_edge_k<<<nE, 256, 0, stream>>>(xnf, rad, wvh, attw, ei, agg, nE);
    cvt_pad_k<<<CDIV((long long)nN * NODE_ELT, 256), 256, 0, stream>>>(
        agg, aggh, nN * KSH, 128, nN * KSH);
    gemm16<0,0,0,0,0,1><<<dim3(CDIV(nN * KSH, 64), 2), 256, 0, stream>>>(
        aggh, 128, woh, 128, nullptr, nullptr, 1, 1, x, nullptr, 128, nN * KSH, 128, 128);

    // --- S2-gated FFN half ---
    rmsnorm_k<<<nN, 128, 0, stream>>>(x, n2, xnf, xnh, nN);
    gemm16<0,1,0,0,1,0><<<dim3(CDIV(nN,64), FFNF/64), 256, 0, stream>>>(
        xnh, NODE_ELT, wgh, FFNF, nullptr, nullptr, 1, 1, gate, nullptr, FFNF, nN, 128, FFNF);
    gemm16<0,0,1,1,0,0><<<dim3(CDIV(nN * KSH, 64), FFNF/64), 256, 0, stream>>>(
        xnh, 128, wf1, FFNF, nullptr, gate, KSH, FFNF, nullptr, hid, FFNF,
        nN * KSH, 128, FFNF);
    gemm16<0,0,0,0,0,1><<<dim3(CDIV(nN * KSH, 64), 2), 256, 0, stream>>>(
        hid, FFNF, wf2, 128, nullptr, nullptr, 1, 1, x, nullptr, 128, nN * KSH, FFNF, 128);
  }

  // ---- final RMS norm, in place on d_out ----
  rmsnorm_k<<<nN, 128, 0, stream>>>(x, nf, x, nullptr, nN);
}
